// Encoder_5179730559754
// MI455X (gfx1250) — matrix operand order repaired and checked by reading, not compiled, not run on hardware
//
#include <hip/hip_runtime.h>
#include <hip/hip_bf16.h>
#include <math.h>

typedef __attribute__((ext_vector_type(16))) _Float16 v16h;
typedef __attribute__((ext_vector_type(8)))  _Float16 v8h;
typedef __attribute__((ext_vector_type(8)))  float    v8f;

#define N_NODES  20000
#define N_EDGES  200000
#define N_GRAPHS 500
#define N_LS     20000

// ---------------------------------------------------------------------------
// WMMA helpers (gfx1250, wave32).  D = A(16x32 f16) * B(32x16 f16) + C(f32)
// ---------------------------------------------------------------------------
static __device__ __forceinline__ v8f wmma_f16(v16h a, v16h b, v8f c) {
  return __builtin_amdgcn_wmma_f32_16x16x32_f16(false, a, false, b, (short)0, c,
                                                false, false);
}

// A tile: 16(M) x 32(K) f16 from LDS row-major [16][ldk] (ldk multiple of 32,
// base 16B aligned).  Two ds_load_b128 per lane.
// 16-bit A layout: lanes 0-15: M=lane, halves 0..7 = K0..7, 8..15 = K16..23;
// lanes 16-31: M=lane-16, halves 0..7 = K8..15, 8..15 = K24..31.
static __device__ __forceinline__ v16h load_A_lds(const _Float16* X, int ldk, int k0) {
  int lane = threadIdx.x & 31;
  int m    = lane & 15;
  int kh   = (lane & 16) ? 8 : 0;
  const _Float16* row = X + m * ldk + k0 + kh;
  v8h lo = *(const v8h*)row;
  v8h hi = *(const v8h*)(row + 16);
  return __builtin_shufflevector(lo, hi, 0, 1, 2, 3, 4, 5, 6, 7,
                                 8, 9, 10, 11, 12, 13, 14, 15);
}

// B tile from pre-packed f16 arena: P[tile][lane][16 halves].
// Two unconditional global_load_b128 per lane, zero VALU.
static __device__ __forceinline__ v16h load_B_packed(const _Float16* __restrict__ P,
                                                     int tile) {
  const _Float16* p = P + ((long)tile << 9) + ((threadIdx.x & 31) << 4);
  v8h lo = *(const v8h*)p;
  v8h hi = *(const v8h*)(p + 8);
  return __builtin_shufflevector(lo, hi, 0, 1, 2, 3, 4, 5, 6, 7,
                                 8, 9, 10, 11, 12, 13, 14, 15);
}

// pack f32 weight W (row-major [*][ldn]) into B-operand layout, zero padded.
// tile = kt*ntiles + nt; lane 0-15: N=nt*16+lane, K=kt*32+{0..7,16..23};
// lanes 16-31: N=nt*16+lane-16, K=kt*32+{8..15,24..31}.
__global__ void pack_B_kernel(const float* __restrict__ W, int ldn, int Klim,
                              int Nlim, int ntiles, int total, _Float16* __restrict__ P) {
  int tid = blockIdx.x * blockDim.x + threadIdx.x;
  if (tid >= total) return;
  int h    = tid & 15;
  int lane = (tid >> 4) & 31;
  int tile = tid >> 9;
  int kt   = tile / ntiles, nt = tile - kt * ntiles;
  int n    = nt * 16 + (lane & 15);
  int k    = kt * 32 + ((lane & 16) >> 1) + (h & 7) + ((h & 8) << 1);
  float v  = (k < Klim && n < Nlim) ? W[(long)k * ldn + n] : 0.f;
  P[tid] = (_Float16)v;
}

static __device__ __forceinline__ void store2h(_Float16* dst, float a, float b) {
  union { _Float16 h[2]; unsigned u; } p;
  p.h[0] = (_Float16)a; p.h[1] = (_Float16)b;
  *(unsigned*)dst = p.u;
}

// ---------------------------------------------------------------------------
__global__ void zero_kernel(float* p, long n) {
  long i = (long)blockIdx.x * blockDim.x + threadIdx.x;
  if (i < n) p[i] = 0.f;
}

// ---------------------------------------------------------------------------
// K1: edge conv.  hid = relu(EA@W1+b1); msg_o = sum_i x[src,i]*(hid@W2[:,32i+o]
//     + b2[32i+o]); atomic scatter into h0[tgt].  One wave / 16 edges.
// ---------------------------------------------------------------------------
__global__ void edge_conv_kernel(const float* __restrict__ x,
                                 const int* __restrict__ edge_index,
                                 const float* __restrict__ edge_attr,
                                 const _Float16* __restrict__ W1p,   // 4 tiles
                                 const float* __restrict__ b1,
                                 const _Float16* __restrict__ W2p,   // 2x64 tiles
                                 const _Float16* __restrict__ b2p,   // 2 tiles
                                 float* __restrict__ h0) {
  __shared__ alignas(16) _Float16 ea16[16][32];
  __shared__ alignas(16) _Float16 x16[16][32];
  __shared__ alignas(16) float    x32[16][32];
  __shared__ alignas(16) _Float16 hid16[16][64];
  __shared__ int tgts[16];
  __shared__ int valid[16];
  int lane = threadIdx.x;
  int e0   = blockIdx.x * 16;
  for (int r = 0; r < 16; ++r) {
    int e = e0 + r; if (e >= N_EDGES) e = N_EDGES - 1;
    int src  = edge_index[e];                  // uniform -> scalar load
    float xv = x[src * 32 + lane];
    x16[r][lane] = (_Float16)xv;
    x32[r][lane] = xv;
    float ea = edge_attr[e * 16 + (lane & 15)];
    if (lane & 16) ea = 0.f;
    ea16[r][lane] = (_Float16)ea;
  }
  {
    int e = e0 + (lane & 15);
    int ok = (e < N_EDGES);
    if (!ok) e = N_EDGES - 1;
    valid[lane & 15] = ok;                     // lanes 16-31 write same value
    tgts[lane & 15]  = edge_index[N_EDGES + e];
  }
  __syncthreads();

  int col  = lane & 15;
  int rofs = (lane & 16) ? 8 : 0;

  // GEMM1: hid = relu(EA @ W1 + b1)   (K=16 zero-padded to 32 in both operands)
  v16h aEA = load_A_lds(&ea16[0][0], 32, 0);
#pragma unroll
  for (int nt = 0; nt < 4; ++nt) {
    v8f  c = {};
    v16h b = load_B_packed(W1p, nt);
    c = wmma_f16(aEA, b, c);
    int   n  = nt * 16 + col;
    float bb = b1[n];
#pragma unroll
    for (int r = 0; r < 8; ++r) {
      float v = c[r] + bb;
      hid16[rofs + r][n] = (_Float16)(v > 0.f ? v : 0.f);
    }
  }
  __syncthreads();

  // msg init = X @ B2mat  (b2 viewed as 32x32)
  v16h aX = load_A_lds(&x16[0][0], 32, 0);
  v8f msg0 = {}, msg1 = {};
  msg0 = wmma_f16(aX, load_B_packed(b2p, 0), msg0);
  msg1 = wmma_f16(aX, load_B_packed(b2p, 1), msg1);

  v16h aH0 = load_A_lds(&hid16[0][0], 64, 0);
  v16h aH1 = load_A_lds(&hid16[0][0], 64, 32);
  for (int i2 = 0; i2 < 16; ++i2) {
    int i = i2 * 2;
    // ew tiles for columns [32i,32i+16), [32i+16,32i+32), and same for i+1
    v8f ew0 = {}, ew1 = {}, ew2 = {}, ew3 = {};
    ew0 = wmma_f16(aH0, load_B_packed(W2p, 2 * i), ew0);
    ew0 = wmma_f16(aH1, load_B_packed(W2p, 64 + 2 * i), ew0);
    ew1 = wmma_f16(aH0, load_B_packed(W2p, 2 * i + 1), ew1);
    ew1 = wmma_f16(aH1, load_B_packed(W2p, 64 + 2 * i + 1), ew1);
    ew2 = wmma_f16(aH0, load_B_packed(W2p, 2 * i + 2), ew2);
    ew2 = wmma_f16(aH1, load_B_packed(W2p, 64 + 2 * i + 2), ew2);
    ew3 = wmma_f16(aH0, load_B_packed(W2p, 2 * i + 3), ew3);
    ew3 = wmma_f16(aH1, load_B_packed(W2p, 64 + 2 * i + 3), ew3);
#pragma unroll
    for (int r = 0; r < 8; ++r) {
      float2 xv = *(const float2*)&x32[rofs + r][i];
      msg0[r] += xv.x * ew0[r] + xv.y * ew2[r];
      msg1[r] += xv.x * ew1[r] + xv.y * ew3[r];
    }
  }
#pragma unroll
  for (int r = 0; r < 8; ++r) {
    int rr = rofs + r;
    if (valid[rr]) {
      int t = tgts[rr];
      atomicAdd(&h0[t * 32 + col],      msg0[r]);
      atomicAdd(&h0[t * 32 + 16 + col], msg1[r]);
    }
  }
}

// ---------------------------------------------------------------------------
// K2: h0 += x @ econv_root + econv_bias
// ---------------------------------------------------------------------------
__global__ void econv_root_kernel(const float* __restrict__ x,
                                  const _Float16* __restrict__ rootp, // 2 tiles
                                  const float* __restrict__ bias,
                                  float* __restrict__ h0) {
  __shared__ alignas(16) _Float16 x16[16][32];
  int lane = threadIdx.x;
  int n0   = blockIdx.x * 16;
  for (int r = 0; r < 16; ++r) {
    int n = n0 + r; if (n >= N_NODES) n = N_NODES - 1;
    x16[r][lane] = (_Float16)x[n * 32 + lane];
  }
  __syncthreads();
  int col = lane & 15, rofs = (lane & 16) ? 8 : 0;
  v16h a = load_A_lds(&x16[0][0], 32, 0);
#pragma unroll
  for (int nt = 0; nt < 2; ++nt) {
    v8f c = {};
    c = wmma_f16(a, load_B_packed(rootp, nt), c);
    int   n  = nt * 16 + col;
    float bb = bias[n];
#pragma unroll
    for (int r = 0; r < 8; ++r) {
      int node = n0 + rofs + r;
      if (node < N_NODES) {
        float* p = &h0[node * 32 + n];
        *p = *p + c[r] + bb;
      }
    }
  }
}

// ---------------------------------------------------------------------------
// GAT: xw = Hin @ W  (ic in {32,64}; out 256 = 4 heads x 64; W packed ktiles x 16)
// ---------------------------------------------------------------------------
__global__ void gat_xw_kernel(const float* __restrict__ Hin, int ic,
                              const _Float16* __restrict__ Wp,
                              float* __restrict__ xw) {
  extern __shared__ v8h st_v[];                  // 16 * ic halves, 16B aligned
  _Float16* st = (_Float16*)st_v;
  int lane = threadIdx.x;
  int n0   = blockIdx.x * 16;
  int cg   = blockIdx.y;                         // column group of 64
  for (int r = 0; r < 16; ++r) {
    int n = n0 + r; if (n >= N_NODES) n = N_NODES - 1;
    if (ic == 32) {
      st[r * 32 + lane] = (_Float16)Hin[n * 32 + lane];
    } else {
      float2 v = ((const float2*)(Hin + (long)n * 64))[lane];
      store2h(&st[r * 64 + 2 * lane], v.x, v.y);
    }
  }
  __syncthreads();
  int col = lane & 15, rofs = (lane & 16) ? 8 : 0;
  int ks  = ic >> 5;
  for (int nt = 0; nt < 4; ++nt) {
    int ntile = cg * 4 + nt;
    v8f c = {};
    for (int k = 0; k < ks; ++k) {
      v16h a = load_A_lds(st, ic, k * 32);
      c = wmma_f16(a, load_B_packed(Wp, k * 16 + ntile), c);
    }
#pragma unroll
    for (int r = 0; r < 8; ++r) {
      int node = n0 + rofs + r;
      if (node < N_NODES) xw[node * 256 + ntile * 16 + col] = c[r];
    }
  }
}

__global__ void gat_a_kernel(const float* __restrict__ xw,
                             const float* __restrict__ asrc,
                             const float* __restrict__ adst,
                             float* an_s, float* an_d) {
  int tid = blockIdx.x * blockDim.x + threadIdx.x;
  if (tid >= N_NODES * 4) return;
  int n = tid >> 2, h = tid & 3;
  const float* p  = xw + n * 256 + h * 64;
  const float* as = asrc + h * 64;
  const float* ad = adst + h * 64;
  float s1 = 0.f, s2 = 0.f;
  for (int o = 0; o < 64; ++o) { float v = p[o]; s1 += v * as[o]; s2 += v * ad[o]; }
  an_s[tid] = s1; an_d[tid] = s2;
}

// sortable-uint encoding for float atomicMax (handles negatives)
static __device__ __forceinline__ unsigned enc_f(float f) {
  unsigned u = __float_as_uint(f);
  return (u & 0x80000000u) ? ~u : (u | 0x80000000u);
}
static __device__ __forceinline__ float dec_f(unsigned u) {
  return __uint_as_float((u & 0x80000000u) ? (u & 0x7FFFFFFFu) : ~u);
}

__global__ void gat_e1_kernel(const int* __restrict__ ei,
                              const float* __restrict__ an_s,
                              const float* __restrict__ an_d,
                              float* ebuf, unsigned* mbuf) {
  int e  = blockIdx.x * blockDim.x + threadIdx.x;
  int nv = N_EDGES + N_NODES;
  if (e >= nv) return;
  int s = (e < N_EDGES) ? ei[e]           : (e - N_EDGES);
  int t = (e < N_EDGES) ? ei[N_EDGES + e] : (e - N_EDGES);
  for (int h = 0; h < 4; ++h) {
    float v = an_s[s * 4 + h] + an_d[t * 4 + h];
    v = (v > 0.f) ? v : 0.2f * v;                       // leaky_relu 0.2
    ebuf[e * 4 + h] = v;
    atomicMax(&mbuf[t * 4 + h], enc_f(v));
  }
}

__global__ void gat_e2_kernel(const int* __restrict__ ei, float* ebuf,
                              const unsigned* __restrict__ mbuf, float* den) {
  int e  = blockIdx.x * blockDim.x + threadIdx.x;
  int nv = N_EDGES + N_NODES;
  if (e >= nv) return;
  int t = (e < N_EDGES) ? ei[N_EDGES + e] : (e - N_EDGES);
  for (int h = 0; h < 4; ++h) {
    float ex = __expf(ebuf[e * 4 + h] - dec_f(mbuf[t * 4 + h]));
    ebuf[e * 4 + h] = ex;
    atomicAdd(&den[t * 4 + h], ex);
  }
}

__global__ void gat_e3_kernel(const int* __restrict__ ei,
                              const float* __restrict__ ebuf,
                              const float* __restrict__ den,
                              const float* __restrict__ xw, float* agg) {
  long tid = (long)blockIdx.x * blockDim.x + threadIdx.x;
  long nv  = (long)(N_EDGES + N_NODES) * 64;
  if (tid >= nv) return;
  int e = (int)(tid >> 6), o = (int)(tid & 63);
  int s = (e < N_EDGES) ? ei[e]           : (e - N_EDGES);
  int t = (e < N_EDGES) ? ei[N_EDGES + e] : (e - N_EDGES);
  for (int h = 0; h < 4; ++h) {
    float alpha = ebuf[e * 4 + h] / den[t * 4 + h];
    atomicAdd(&agg[t * 256 + h * 64 + o], alpha * xw[s * 256 + h * 64 + o]);
  }
}

__global__ void gat_mean_kernel(const float* __restrict__ agg,
                                const float* __restrict__ b, float* Hout) {
  int tid = blockIdx.x * blockDim.x + threadIdx.x;
  if (tid >= N_NODES * 64) return;
  int n = tid >> 6, o = tid & 63;
  float s = 0.f;
  for (int h = 0; h < 4; ++h) s += agg[n * 256 + h * 64 + o];
  Hout[tid] = 0.25f * s + b[o];
}

// ---------------------------------------------------------------------------
// Generic WMMA MLP: gather features -> relu(X@W1+b1) -> @W2+b2 -> pool/rows.
// Hidden = 128.  One wave / 16 rows.  Weights pre-packed f16.
// scal_mode: 1=raw scalar (1 feat), 2=cos/sin (2 feats); tail only if Kpad>segLim
// epi_mode:  0=pool: dst[gmap[pool_idx[row]]*stride + colOff + col] += v
//            1=rows: dst[row*stride + col] += v
// ---------------------------------------------------------------------------
__global__ void mlp_kernel(const float* __restrict__ Hmat,
                           const int* idx0, const int* idx1,
                           const int* idx2, const int* idx3, int n_idx,
                           const float* scal0, int scal_mode,
                           const _Float16* __restrict__ W1p,  // ks1 x 8 tiles
                           const float* __restrict__ b1,
                           const _Float16* __restrict__ W2p,  // 4 x outTiles tiles
                           const float* __restrict__ b2,
                           int K, int Kpad, int outDim, int nrows,
                           int epi_mode, const int* pool_idx, const int* gmap,
                           float* dst, int dstStride, int colOff) {
  extern __shared__ v8h X_v[];                  // 16*Kpad + 16*128 halves
  _Float16* X  = (_Float16*)X_v;
  _Float16* Hl = X + 16 * Kpad;
  int lane   = threadIdx.x;
  int row0   = blockIdx.x * 16;
  int segLim = n_idx * 64;
  __builtin_prefetch(W1p + (lane << 4), 0, 1);
  // --- gather: H segments, one float2 global load + one b32 LDS store / lane
  for (int u = 0; u < 16 * n_idx; ++u) {
    int r = u / n_idx, seg = u - r * n_idx;
    int row = row0 + r; if (row >= nrows) row = nrows - 1;
    const int* ip = (seg == 0) ? idx0 : (seg == 1) ? idx1 : (seg == 2) ? idx2 : idx3;
    int hrow = ip[row];                                   // uniform -> scalar load
    float2 v = ((const float2*)(Hmat + (long)hrow * 64))[lane];
    store2h(&X[r * Kpad + seg * 64 + 2 * lane], v.x, v.y);
  }
  // --- gather: scalar features + zero tail (tail is 0 or 32 wide)
  int padTail = Kpad - segLim;
  if (padTail) {
    for (int t = lane; t < 16 * 32; t += 32) {
      int r = t >> 5, f = t & 31;
      int row = row0 + r; if (row >= nrows) row = nrows - 1;
      float v = 0.f;
      if (scal_mode == 1)      { if (f == 0) v = scal0[row]; }
      else if (scal_mode == 2) { if (f == 0) v = __cosf(scal0[row]);
                                 else if (f == 1) v = __sinf(scal0[row]); }
      X[r * Kpad + segLim + f] = (_Float16)v;
    }
  }
  __syncthreads();

  int col = lane & 15, rofs = (lane & 16) ? 8 : 0;
  int ks1 = Kpad >> 5;
  // hidden = relu(X @ W1 + b1)
  for (int nt = 0; nt < 8; ++nt) {
    v8f c = {};
    for (int k = 0; k < ks1; ++k) {
      v16h a = load_A_lds(X, Kpad, k * 32);
      c = wmma_f16(a, load_B_packed(W1p, k * 8 + nt), c);
    }
    int   n  = nt * 16 + col;
    float bb = b1[n];
#pragma unroll
    for (int r = 0; r < 8; ++r) {
      float v = c[r] + bb;
      Hl[(rofs + r) * 128 + n] = (_Float16)(v > 0.f ? v : 0.f);
    }
  }
  __syncthreads();

  int outTiles = (outDim + 15) >> 4;
  for (int nt = 0; nt < outTiles; ++nt) {
    v8f c = {};
    for (int k = 0; k < 4; ++k) {
      v16h a = load_A_lds(Hl, 128, k * 32);
      c = wmma_f16(a, load_B_packed(W2p, k * outTiles + nt), c);
    }
    int n = nt * 16 + col;
    if (n < outDim) {
      float bb = b2[n];
#pragma unroll
      for (int r = 0; r < 8; ++r) {
        int row = row0 + rofs + r;
        if (row < nrows) {
          float v = c[r] + bb;
          if (epi_mode == 0) {
            int g = gmap[pool_idx[row]];
            atomicAdd(&dst[g * dstStride + colOff + n], v);
          } else {
            atomicAdd(&dst[row * dstStride + n], v);
          }
        }
      }
    }
  }
}

// ---------------------------------------------------------------------------
// psi rotation / sigmoid gate / LS pooling, then radii.
// ---------------------------------------------------------------------------
__global__ void psi_combine_kernel(const float* __restrict__ c_rows,
                                   const float* __restrict__ ps_rows,
                                   const float* __restrict__ psis,
                                   const int* __restrict__ LS_map, float* LS) {
  int i = blockIdx.x * blockDim.x + threadIdx.x;
  if (i >= 160000) return;
  float cc = c_rows[i];
  float px = ps_rows[2 * i], py = ps_rows[2 * i + 1];
  float nr = sqrtf(px * px + py * py);
  float nd = fmaxf(nr, 1e-12f);
  float pc = px / nd, pn = py / nd;
  float nc = 1.f / (1.f + __expf(-cc));
  float cp = __cosf(psis[i]), sp = __sinf(psis[i]);
  float sx = (cp * pc - sp * pn) * nc;
  float sy = (sp * pc + cp * pn) * nc;
  int g = LS_map[i];
  atomicAdd(&LS[2 * g],     sx);
  atomicAdd(&LS[2 * g + 1], sy);
}

__global__ void radii_kernel(const float* __restrict__ LS, float* radii) {
  int i = blockIdx.x * blockDim.x + threadIdx.x;
  if (i >= N_LS) return;
  float a = LS[2 * i], b = LS[2 * i + 1];
  radii[i] = sqrtf(a * a + b * b);
}

// ---------------------------------------------------------------------------
// Host launcher
// ---------------------------------------------------------------------------
extern "C" void kernel_launch(void* const* d_in, const int* in_sizes, int n_in,
                              void* d_out, int out_size, void* d_ws, size_t ws_size,
                              hipStream_t stream) {
  // params flattened in jax pytree (sorted dict key) order, then other inputs.
  const float* D_W1       = (const float*)d_in[0];
  const float* D_W2       = (const float*)d_in[1];
  const float* D_b1       = (const float*)d_in[2];
  const float* D_b2       = (const float*)d_in[3];
  const float* al_W1      = (const float*)d_in[4];
  const float* al_W2      = (const float*)d_in[5];
  const float* al_b1      = (const float*)d_in[6];
  const float* al_b2      = (const float*)d_in[7];
  const float* c_W1       = (const float*)d_in[8];
  const float* c_W2       = (const float*)d_in[9];
  const float* c_b1       = (const float*)d_in[10];
  const float* c_b2       = (const float*)d_in[11];
  const float* econv_bias = (const float*)d_in[12];
  const float* econv_root = (const float*)d_in[13];
  const float* emlp_W1    = (const float*)d_in[14];
  const float* emlp_W2    = (const float*)d_in[15];
  const float* emlp_b1    = (const float*)d_in[16];
  const float* emlp_b2    = (const float*)d_in[17];
  const float* g0_W       = (const float*)d_in[18];
  const float* g0_adst    = (const float*)d_in[19];
  const float* g0_asrc    = (const float*)d_in[20];
  const float* g0_b       = (const float*)d_in[21];
  const float* g1_W       = (const float*)d_in[22];
  const float* g1_adst    = (const float*)d_in[23];
  const float* g1_asrc    = (const float*)d_in[24];
  const float* g1_b       = (const float*)d_in[25];
  const float* phi_W1     = (const float*)d_in[26];
  const float* phi_W2     = (const float*)d_in[27];
  const float* phi_b1     = (const float*)d_in[28];
  const float* phi_b2     = (const float*)d_in[29];
  const float* ss_W1      = (const float*)d_in[30];
  const float* ss_W2      = (const float*)d_in[31];
  const float* ss_b1      = (const float*)d_in[32];
  const float* ss_b2      = (const float*)d_in[33];
  const float* x          = (const float*)d_in[34];
  const int*   edge_index = (const int*)d_in[35];
  const float* edge_attr  = (const float*)d_in[36];
  const float* distances  = (const float*)d_in[37];
  const int*   dist_idx   = (const int*)d_in[38];
  const float* phis       = (const float*)d_in[39];
  const int*   phi_idx    = (const int*)d_in[40];
  const float* psis       = (const float*)d_in[41];
  const int*   psi_idx    = (const int*)d_in[42];
  const int*   node_map   = (const int*)d_in[43];
  const int*   LS_map     = (const int*)d_in[44];
  const int*   alpha_idx  = (const int*)d_in[45];
  float* out = (float*)d_out;

  float* ws    = (float*)d_ws;
  float* h0    = ws;                  // 640000
  float* h1    = h0 + 640000;         // 1,280,000
  float* Hf    = h1 + 1280000;        // 1,280,000  (offset 256B aligned)
  float* xw    = Hf + 1280000;        // 5,120,000
  float* agg   = xw + 5120000;        // 5,120,000
  float* an_s  = agg + 5120000;       // 80,000
  float* an_d  = an_s + 80000;        // 80,000
  unsigned* mmax = (unsigned*)(an_d + 80000); // 80,000
  float* den   = an_d + 160000;       // 80,000
  float* ebuf  = den + 80000;         // 880,000
  float* crows = ebuf + 880000;       // 160,000
  float* prows = crows + 160000;      // 320,000
  float* LS    = prows + 320000;      // 40,000
  float* radii = LS + 40000;          // 20,000
  // packed f16 weight arenas (each a multiple of 512 halves -> 32B aligned)
  _Float16* pk      = (_Float16*)(radii + 20000);
  _Float16* emlpW1p = pk;                   //   4*512
  _Float16* emlpW2p = emlpW1p + 2048;       // 128*512
  _Float16* b2p     = emlpW2p + 65536;      //   2*512
  _Float16* rootp   = b2p + 1024;           //   2*512
  _Float16* g0Wp    = rootp + 1024;         //  16*512
  _Float16* g1Wp    = g0Wp + 8192;          //  32*512
  _Float16* DW1p    = g1Wp + 16384;         //  40*512
  _Float16* DW2p    = DW1p + 20480;         //   8*512
  _Float16* alW1p   = DW2p + 4096;          //  40*512
  _Float16* alW2p   = alW1p + 20480;        //   8*512
  _Float16* phiW1p  = alW2p + 4096;         //  56*512
  _Float16* phiW2p  = phiW1p + 28672;       //   8*512
  _Float16* cW1p    = phiW2p + 4096;        //  64*512
  _Float16* cW2p    = cW1p + 32768;         //   4*512
  _Float16* ssW1p   = cW2p + 2048;          //  64*512
  _Float16* ssW2p   = ssW1p + 32768;        //   4*512

  auto zero = [&](float* p, long n) {
    zero_kernel<<<dim3((unsigned)((n + 255) / 256)), dim3(256), 0, stream>>>(p, n);
  };
  auto pack = [&](const float* W, int ldn, int Klim, int Nlim, int ktiles,
                  int ntiles, _Float16* P) {
    int total = ktiles * ntiles * 512;
    pack_B_kernel<<<(total + 255) / 256, 256, 0, stream>>>(W, ldn, Klim, Nlim,
                                                           ntiles, total, P);
  };

  // 0) pack all weights to f16 B-operand layout (tiny, L2-resident)
  pack(emlp_W1,   64,  16,   64, 1,  4, emlpW1p);
  pack(emlp_W2, 1024,  64, 1024, 2, 64, emlpW2p);
  pack(emlp_b2,   32,  32,   32, 1,  2, b2p);
  pack(econv_root,32,  32,   32, 1,  2, rootp);
  pack(g0_W,     256,  32,  256, 1, 16, g0Wp);
  pack(g1_W,     256,  64,  256, 2, 16, g1Wp);
  pack(D_W1,     128, 129,  128, 5,  8, DW1p);
  pack(D_W2,      32, 128,   32, 4,  2, DW2p);
  pack(al_W1,    128, 129,  128, 5,  8, alW1p);
  pack(al_W2,     32, 128,   32, 4,  2, alW2p);
  pack(phi_W1,   128, 194,  128, 7,  8, phiW1p);
  pack(phi_W2,    32, 128,   32, 4,  2, phiW2p);
  pack(c_W1,     128, 256,  128, 8,  8, cW1p);
  pack(c_W2,       1, 128,    1, 4,  1, cW2p);
  pack(ss_W1,    128, 256,  128, 8,  8, ssW1p);
  pack(ss_W2,      2, 128,    2, 4,  1, ssW2p);

  zero(h0, 640000);
  zero(out, (long)out_size);

  // 1) edge conv + root
  edge_conv_kernel<<<12500, 32, 0, stream>>>(x, edge_index, edge_attr,
                                             emlpW1p, emlp_b1, emlpW2p, b2p, h0);
  econv_root_kernel<<<1250, 32, 0, stream>>>(x, rootp, econv_bias, h0);

  // 2) two GAT layers
  const int nv = N_EDGES + N_NODES;
  for (int li = 0; li < 2; ++li) {
    const float*    Hin  = (li == 0) ? h0 : h1;
    int             ic   = (li == 0) ? 32 : 64;
    const _Float16* Wp   = (li == 0) ? g0Wp : g1Wp;
    const float*    as_  = (li == 0) ? g0_asrc : g1_asrc;
    const float*    ad_  = (li == 0) ? g0_adst : g1_adst;
    const float*    bb   = (li == 0) ? g0_b : g1_b;
    float*          Hout = (li == 0) ? h1 : Hf;
    size_t sh = (size_t)16 * ic * sizeof(_Float16);
    gat_xw_kernel<<<dim3(1250, 4), 32, sh, stream>>>(Hin, ic, Wp, xw);
    gat_a_kernel<<<(N_NODES * 4 + 255) / 256, 256, 0, stream>>>(xw, as_, ad_, an_s, an_d);
    zero((float*)mmax, 80000);
    zero(den, 80000);
    zero(agg, 5120000);
    gat_e1_kernel<<<(nv + 255) / 256, 256, 0, stream>>>(edge_index, an_s, an_d, ebuf, mmax);
    gat_e2_kernel<<<(nv + 255) / 256, 256, 0, stream>>>(edge_index, ebuf, mmax, den);
    long nv64 = (long)nv * 64;
    gat_e3_kernel<<<(unsigned)((nv64 + 255) / 256), 256, 0, stream>>>(edge_index, ebuf,
                                                                      den, xw, agg);
    gat_mean_kernel<<<(N_NODES * 64 + 255) / 256, 256, 0, stream>>>(agg, bb, Hout);
  }

  // 3) D task (both directions) -> out cols [0,32)
  {
    size_t sh = (size_t)(16 * 160 + 16 * 128) * sizeof(_Float16);
    mlp_kernel<<<5000, 32, sh, stream>>>(Hf, dist_idx, dist_idx + 80000, nullptr, nullptr, 2,
        distances, 1, DW1p, D_b1, DW2p, D_b2, 129, 160, 32, 80000,
        0, dist_idx, node_map, out, 96, 0);
    mlp_kernel<<<5000, 32, sh, stream>>>(Hf, dist_idx + 80000, dist_idx, nullptr, nullptr, 2,
        distances, 1, DW1p, D_b1, DW2p, D_b2, 129, 160, 32, 80000,
        0, dist_idx, node_map, out, 96, 0);
  }
  // 4) phi task -> out cols [32,64)
  {
    size_t sh = (size_t)(16 * 224 + 16 * 128) * sizeof(_Float16);
    mlp_kernel<<<10000, 32, sh, stream>>>(Hf, phi_idx, phi_idx + 160000, phi_idx + 320000,
        nullptr, 3, phis, 2, phiW1p, phi_b1, phiW2p, phi_b2, 194, 224, 32, 160000,
        0, phi_idx, node_map, out, 96, 32);
    mlp_kernel<<<10000, 32, sh, stream>>>(Hf, phi_idx + 320000, phi_idx + 160000, phi_idx,
        nullptr, 3, phis, 2, phiW1p, phi_b1, phiW2p, phi_b2, 194, 224, 32, 160000,
        0, phi_idx, node_map, out, 96, 32);
  }
  // 5) c & ss MLPs -> per-row, then psi combine -> LS -> radii
  zero(crows, 160000); zero(prows, 320000); zero(LS, 40000);
  {
    size_t sh = (size_t)(16 * 256 + 16 * 128) * sizeof(_Float16);
    const int* i0 = psi_idx;
    const int* i1 = psi_idx + 160000;
    const int* i2 = psi_idx + 320000;
    const int* i3 = psi_idx + 480000;
    mlp_kernel<<<10000, 32, sh, stream>>>(Hf, i0, i1, i2, i3, 4, nullptr, 0,
        cW1p, c_b1, cW2p, c_b2, 256, 256, 1, 160000, 1, nullptr, nullptr, crows, 1, 0);
    mlp_kernel<<<10000, 32, sh, stream>>>(Hf, i3, i2, i1, i0, 4, nullptr, 0,
        cW1p, c_b1, cW2p, c_b2, 256, 256, 1, 160000, 1, nullptr, nullptr, crows, 1, 0);
    mlp_kernel<<<10000, 32, sh, stream>>>(Hf, i0, i1, i2, i3, 4, nullptr, 0,
        ssW1p, ss_b1, ssW2p, ss_b2, 256, 256, 2, 160000, 1, nullptr, nullptr, prows, 2, 0);
    mlp_kernel<<<10000, 32, sh, stream>>>(Hf, i3, i2, i1, i0, 4, nullptr, 0,
        ssW1p, ss_b1, ssW2p, ss_b2, 256, 256, 2, 160000, 1, nullptr, nullptr, prows, 2, 0);
  }
  psi_combine_kernel<<<(160000 + 255) / 256, 256, 0, stream>>>(crows, prows, psis, LS_map, LS);
  radii_kernel<<<(N_LS + 255) / 256, 256, 0, stream>>>(LS, radii);

  // 6) alpha task -> out cols [64,96)
  {
    size_t sh = (size_t)(16 * 160 + 16 * 128) * sizeof(_Float16);
    mlp_kernel<<<1250, 32, sh, stream>>>(Hf, alpha_idx, alpha_idx + 20000, nullptr, nullptr, 2,
        radii, 1, alW1p, al_b1, alW2p, al_b2, 129, 160, 32, 20000,
        0, alpha_idx, node_map, out, 96, 64);
    mlp_kernel<<<1250, 32, sh, stream>>>(Hf, alpha_idx + 20000, alpha_idx, nullptr, nullptr, 2,
        radii, 1, alW1p, al_b1, alW2p, al_b2, 129, 160, 32, 20000,
        0, alpha_idx, node_map, out, 96, 64);
  }
}
